// RGAT_80393197846975
// MI455X (gfx1250) — matrix operand order repaired and checked by reading, not compiled, not run on hardware
//
#include <hip/hip_runtime.h>
#include <cstdint>
#include <cstddef>

// ---------------- problem constants (from reference) ----------------
#define NP_   60000
#define NA_   30000
#define EC_   300000
#define EW_   150000
#define EB_   150000
#define INF_  1024
#define HID_  512
#define H_    4
#define D_    128
#define NCLS_ 153

typedef __attribute__((ext_vector_type(16))) __bf16 v16bf;
typedef __attribute__((ext_vector_type(8)))  float  v8f;

// ---------------- small device helpers ----------------
__device__ __forceinline__ unsigned short f2bf(float f) {
  unsigned u = __float_as_uint(f);
  u += 0x7FFFu + ((u >> 16) & 1u);          // round-to-nearest-even
  return (unsigned short)(u >> 16);
}
__device__ __forceinline__ unsigned fenc(float f) {   // order-preserving f32->u32
  unsigned u = __float_as_uint(f);
  return (u & 0x80000000u) ? ~u : (u | 0x80000000u);
}
__device__ __forceinline__ float fdec(unsigned k) {
  return (k & 0x80000000u) ? __uint_as_float(k & 0x7FFFFFFFu)
                           : __uint_as_float(~k);
}
__device__ __forceinline__ void atomAddF(float* p, float v) {
  __hip_atomic_fetch_add(p, v, __ATOMIC_RELAXED, __HIP_MEMORY_SCOPE_AGENT);
}

// ---------------- fill / elementwise ----------------
__global__ void fill_u32_k(unsigned* __restrict__ p, unsigned v, size_t n) {
  size_t i = (size_t)blockIdx.x * blockDim.x + threadIdx.x;
  if (i < n) p[i] = v;
}
__global__ void relu_copy_k(const float* __restrict__ a, float* __restrict__ b, size_t n) {
  size_t i = (size_t)blockIdx.x * blockDim.x + threadIdx.x;
  if (i < n) { float v = a[i]; b[i] = v > 0.f ? v : 0.f; }
}

// ---------------- WMMA bf16 GEMM: C(MxN,f32) = A(MxK,f32) @ B(KxN,f32) ----------------
// Block tile 128x128, BK=32, 8 waves (4x2), each wave 32x64 (2x4 WMMA tiles).
// Software pipelined: global loads of tile k+1 overlap WMMAs of tile k.
#define BM 128
#define BN 128
#define BK 32

__global__ __launch_bounds__(256) void gemm_wmma_k(
    const float* __restrict__ A, const float* __restrict__ B,
    float* __restrict__ C, int M, int N, int K,
    const float* __restrict__ bias) {
  __shared__ unsigned short As[BM][BK + 2];   // stride 34 shorts (17-bank) -> conflict free
  __shared__ unsigned short Bs[BN][BK + 2];   // stored transposed: [n][k]

  const int tid  = threadIdx.x;
  const int lane = tid & 31;
  const int wave = tid >> 5;
  const int wm   = wave & 3;        // 0..3 -> M direction (32 rows each)
  const int wn   = wave >> 2;       // 0..1 -> N direction (64 cols each)
  const int m0   = blockIdx.y * BM;
  const int n0   = blockIdx.x * BN;

  // staging registers (double buffer in regs, single buffer in LDS)
  const int arow = tid >> 1;              // 0..127
  const int acol = (tid & 1) * 16;        // 0 or 16
  const int brow = tid >> 3;              // 0..31
  const int bcol = (tid & 7) * 16;        // 0..112
  float regA[16], regB[16];

  auto loadA = [&](int kk) {
    int grow = m0 + arow;
    const float* ap = A + (size_t)grow * K + kk + acol;
    bool ok = (grow < M) && (kk + acol < K);
#pragma unroll
    for (int j = 0; j < 16; j += 4) {
      float4 v = make_float4(0.f, 0.f, 0.f, 0.f);
      if (ok) v = *(const float4*)(ap + j);
      regA[j + 0] = v.x; regA[j + 1] = v.y; regA[j + 2] = v.z; regA[j + 3] = v.w;
    }
    if (ok && (kk + 2 * BK + acol) < K) __builtin_prefetch(ap + 2 * BK, 0, 1);
  };
  auto loadB = [&](int kk) {
    int gk = kk + brow;
    const float* bp = B + (size_t)gk * N + n0 + bcol;
    bool kok = gk < K;
#pragma unroll
    for (int j = 0; j < 16; j += 4) {
      if (kok && (n0 + bcol + j + 3) < N) {
        float4 v = *(const float4*)(bp + j);
        regB[j + 0] = v.x; regB[j + 1] = v.y; regB[j + 2] = v.z; regB[j + 3] = v.w;
      } else {
#pragma unroll
        for (int t = 0; t < 4; ++t) {
          int gc = n0 + bcol + j + t;
          regB[j + t] = (kok && gc < N) ? bp[j + t] : 0.f;
        }
      }
    }
  };
  auto stageLDS = [&]() {
#pragma unroll
    for (int j = 0; j < 16; ++j) As[arow][acol + j] = f2bf(regA[j]);
#pragma unroll
    for (int j = 0; j < 16; ++j) Bs[bcol + j][brow] = f2bf(regB[j]);
  };

  v8f zv = {};
  v8f acc[2][4];
#pragma unroll
  for (int i = 0; i < 2; ++i)
#pragma unroll
    for (int j = 0; j < 4; ++j) acc[i][j] = zv;

  union Frag { v16bf v; unsigned u[8]; };

  loadA(0); loadB(0);

  for (int k0 = 0; k0 < K; k0 += BK) {
    stageLDS();
    __syncthreads();
    // prefetch next K tile into registers while we compute
    if (k0 + BK < K) { loadA(k0 + BK); loadB(k0 + BK); }

    // ---- build fragments per ISA bf16 layouts & issue 8 WMMAs ----
    const int lhalf = lane & 16;
    Frag af[2], bfr[4];
    {
      const int ka = lhalf ? 8 : 0;     // A lanes 0-15: K 0..7 / 16..23 ; lanes 16-31: 8..15 / 24..31
      const int kb = lhalf ? 24 : 16;
#pragma unroll
      for (int ti = 0; ti < 2; ++ti) {
        const unsigned short* base = &As[wm * 32 + ti * 16 + (lane & 15)][0];
#pragma unroll
        for (int j = 0; j < 4; ++j) af[ti].u[j]     = *(const unsigned*)(base + ka + 2 * j);
#pragma unroll
        for (int j = 0; j < 4; ++j) af[ti].u[4 + j] = *(const unsigned*)(base + kb + 2 * j);
      }
      const int kc = lhalf ? 8 : 0;     // B lanes 0-15: K 0..7 / 16..23 ; lanes 16-31: 8..15 / 24..31
#pragma unroll
      for (int tj = 0; tj < 4; ++tj) {
        const unsigned short* base = &Bs[wn * 64 + tj * 16 + (lane & 15)][kc];
#pragma unroll
        for (int j = 0; j < 8; ++j) bfr[tj].u[j] = *(const unsigned*)(base + (j >> 2) * 16 + 2 * (j & 3));
      }
    }
#pragma unroll
    for (int ti = 0; ti < 2; ++ti)
#pragma unroll
      for (int tj = 0; tj < 4; ++tj)
        acc[ti][tj] = __builtin_amdgcn_wmma_f32_16x16x32_bf16(
            false, af[ti].v, false, bfr[tj].v, (short)0, acc[ti][tj], false, false);

    __syncthreads();   // protect LDS before next stage overwrites it
  }

  // ---- epilogue: C/D layout -> lanes 0-15: M=r, lanes 16-31: M=r+8 ----
#pragma unroll
  for (int ti = 0; ti < 2; ++ti) {
#pragma unroll
    for (int tj = 0; tj < 4; ++tj) {
      int row = m0 + wm * 32 + ti * 16 + ((lane & 16) ? 8 : 0);
      int col = n0 + wn * 64 + tj * 16 + (lane & 15);
      if (col < N) {
        float badd = bias ? bias[col] : 0.f;
#pragma unroll
        for (int r = 0; r < 8; ++r) {
          int gr = row + r;
          if (gr < M) C[(size_t)gr * N + col] = acc[ti][tj][r] + badd;
        }
      }
    }
  }
}

// ---------------- wl[h][k] = sum_d W[k, h*D+d] * a[h*D+d] ----------------
__global__ void make_wvec_k(const float* __restrict__ W, const float* __restrict__ a,
                            float* __restrict__ wv, int K) {
  int idx = blockIdx.x * blockDim.x + threadIdx.x;   // H_*K
  if (idx >= H_ * K) return;
  int h = idx / K, k = idx % K;
  const float* wrow = W + (size_t)k * HID_ + h * D_;
  const float* av   = a + h * D_;
  float s = 0.f;
  for (int d = 0; d < D_; ++d) s += wrow[d] * av[d];
  wv[idx] = s;
}

// ---------------- out[n,h] = h[n,:] . wv[h,:]  (one wave per (n,h)) ----------------
__global__ void rowdot_k(const float* __restrict__ h, const float* __restrict__ wv,
                         float* __restrict__ out, int n, int K) {
  int lane = threadIdx.x & 31;
  int pair = blockIdx.x * (blockDim.x >> 5) + (threadIdx.x >> 5);
  if (pair >= n * H_) return;
  int node = pair >> 2, hh = pair & 3;
  const float* row = h + (size_t)node * K;
  const float* w   = wv + (size_t)hh * K;
  float s = 0.f;
  for (int k = lane; k < K; k += 32) s += row[k] * w[k];
#pragma unroll
  for (int m = 16; m >= 1; m >>= 1) s += __shfl_xor(s, m, 32);
  if (lane == 0) out[pair] = s;
}

// ---------------- edge softmax pieces ----------------
__global__ void edge_max_k(const float* __restrict__ el, const float* __restrict__ er,
                           const int* __restrict__ src, const int* __restrict__ dst,
                           int E, unsigned* __restrict__ menc) {
  int gid = blockIdx.x * blockDim.x + threadIdx.x;
  if (gid >= E * H_) return;
  int e = gid >> 2, h = gid & 3;
  float v = el[(size_t)src[e] * H_ + h] + er[(size_t)dst[e] * H_ + h];
  v = v > 0.f ? v : 0.2f * v;                         // leaky_relu(0.2)
  atomicMax(&menc[(size_t)dst[e] * H_ + h], fenc(v));
}

__global__ void edge_expsum_k(const float* __restrict__ el, const float* __restrict__ er,
                              const int* __restrict__ src, const int* __restrict__ dst,
                              int E, const unsigned* __restrict__ menc,
                              float* __restrict__ ee, float* __restrict__ s) {
  int gid = blockIdx.x * blockDim.x + threadIdx.x;
  if (gid >= E * H_) return;
  int e = gid >> 2, h = gid & 3;
  size_t nb = (size_t)dst[e] * H_ + h;
  float v = el[(size_t)src[e] * H_ + h] + er[nb];
  v = v > 0.f ? v : 0.2f * v;
  float x = __expf(v - fdec(menc[nb]));
  ee[(size_t)e * H_ + h] = x;
  atomAddF(&s[nb], x);
}

// ---------------- weighted aggregation: acc[dst] += a * zs[src] ----------------
__global__ void edge_aggr_k(const float* __restrict__ ee, const float* __restrict__ s,
                            const int* __restrict__ src, const int* __restrict__ dst,
                            int E, const float* __restrict__ zs, float* __restrict__ acc) {
  long long gid = (long long)blockIdx.x * blockDim.x + threadIdx.x;
  if (gid >= (long long)E * D_) return;
  int e = (int)(gid >> 7);                // /128
  int d = (int)(gid & 127);
  size_t sb = (size_t)src[e] * HID_;
  size_t db = (size_t)dst[e] * HID_;
  size_t eb = (size_t)e * H_;
  size_t nb = (size_t)dst[e] * H_;
#pragma unroll
  for (int h = 0; h < H_; ++h) {
    float a = ee[eb + h] / (s[nb + h] + 1e-9f);
    int c = h * D_ + d;
    atomAddF(&acc[db + c], a * zs[sb + c]);
  }
}

// ---------------- host orchestration ----------------
static inline int cdiv(long long a, long long b) { return (int)((a + b - 1) / b); }

extern "C" void kernel_launch(void* const* d_in, const int* in_sizes, int n_in,
                              void* d_out, int out_size, void* d_ws, size_t ws_size,
                              hipStream_t stream) {
  (void)in_sizes; (void)n_in; (void)out_size; (void)ws_size;
  const float* x_paper = (const float*)d_in[0];
  const float* x_author= (const float*)d_in[1];
  const int* cites_src = (const int*)d_in[2];
  const int* cites_dst = (const int*)d_in[3];
  const int* writes_src= (const int*)d_in[4];
  const int* writes_dst= (const int*)d_in[5];
  const int* wb_src    = (const int*)d_in[6];
  const int* wb_dst    = (const int*)d_in[7];
  const float* W0 = (const float*)d_in[8];
  const float* al0= (const float*)d_in[9];
  const float* ar0= (const float*)d_in[10];
  const float* W1 = (const float*)d_in[11];
  const float* al1= (const float*)d_in[12];
  const float* ar1= (const float*)d_in[13];
  const float* W2 = (const float*)d_in[14];
  const float* al2= (const float*)d_in[15];
  const float* ar2= (const float*)d_in[16];
  const float* linW = (const float*)d_in[17];
  const float* linb = (const float*)d_in[18];

  char* ws = (char*)d_ws;
  size_t off = 0;
  auto carve = [&](size_t bytes) -> char* {
    char* p = ws + off;
    off += (bytes + 255) & ~(size_t)255;
    return p;
  };
  float*    hpB  = (float*)carve((size_t)NP_ * HID_ * 4);
  float*    haB  = (float*)carve((size_t)NA_ * HID_ * 4);
  float*    zS   = (float*)carve((size_t)NP_ * HID_ * 4);
  float*    accP = (float*)carve((size_t)NP_ * HID_ * 4);
  float*    accA = (float*)carve((size_t)NA_ * HID_ * 4);
  float*    wl   = (float*)carve((size_t)H_ * INF_ * 4);
  float*    wr   = (float*)carve((size_t)H_ * INF_ * 4);
  float*    el   = (float*)carve((size_t)NP_ * H_ * 4);
  float*    er   = (float*)carve((size_t)NP_ * H_ * 4);
  unsigned* menc = (unsigned*)carve((size_t)NP_ * H_ * 4);
  float*    ssum = (float*)carve((size_t)NP_ * H_ * 4);
  float*    ee   = (float*)carve((size_t)EC_ * H_ * 4);

  auto fill = [&](unsigned* p, unsigned v, size_t n) {
    fill_u32_k<<<cdiv((long long)n, 256), 256, 0, stream>>>(p, v, n);
  };
  auto gemm = [&](const float* A, const float* B, float* C, int M, int N, int K,
                  const float* bias) {
    dim3 g(cdiv(N, BN), cdiv(M, BM));
    gemm_wmma_k<<<g, 256, 0, stream>>>(A, B, C, M, N, K, bias);
  };

  // one edge-type GAT conv, accumulating into `acc`
  auto gat_type = [&](const float* h_src, int n_src, const float* h_dst, int n_dst,
                      const int* esrc, const int* edst, int E,
                      const float* W, const float* al, const float* ar, int K,
                      float* acc) {
    // src projection (needed for messages): zS = h_src @ W   [n_src x 512]
    gemm(h_src, W, zS, n_src, HID_, K, nullptr);
    // folded attention vectors: wl/wr = W . al / W . ar   [H x K]
    make_wvec_k<<<cdiv((long long)H_ * K, 256), 256, 0, stream>>>(W, al, wl, K);
    make_wvec_k<<<cdiv((long long)H_ * K, 256), 256, 0, stream>>>(W, ar, wr, K);
    // el = h_src @ wl, er = h_dst @ wr
    rowdot_k<<<cdiv((long long)n_src * H_, 8), 256, 0, stream>>>(h_src, wl, el, n_src, K);
    rowdot_k<<<cdiv((long long)n_dst * H_, 8), 256, 0, stream>>>(h_dst, wr, er, n_dst, K);
    // segment max (encoded), then exp + segment sum
    fill(menc, 0x00800000u /* fenc(-FLT_MAX) */, (size_t)n_dst * H_);
    edge_max_k<<<cdiv((long long)E * H_, 256), 256, 0, stream>>>(el, er, esrc, edst, E, menc);
    fill((unsigned*)ssum, 0u, (size_t)n_dst * H_);
    edge_expsum_k<<<cdiv((long long)E * H_, 256), 256, 0, stream>>>(el, er, esrc, edst, E,
                                                                    menc, ee, ssum);
    // weighted aggregation into acc
    edge_aggr_k<<<cdiv((long long)E * D_, 256), 256, 0, stream>>>(ee, ssum, esrc, edst, E,
                                                                  zS, acc);
  };

  const float* Ws[3]  = {W0, W1, W2};
  const float* als[3] = {al0, al1, al2};
  const float* ars[3] = {ar0, ar1, ar2};

  for (int l = 0; l < 3; ++l) {
    const int K = (l == 0) ? INF_ : HID_;
    const float* hpIn = (l == 0) ? x_paper  : hpB;
    const float* haIn = (l == 0) ? x_author : haB;
    const float* W  = Ws[l];
    const float* al = als[l];
    const float* ar = ars[l];

    fill((unsigned*)accP, 0u, (size_t)NP_ * HID_);
    fill((unsigned*)accA, 0u, (size_t)NA_ * HID_);

    // cites: paper -> paper (type 0)
    gat_type(hpIn, NP_, hpIn, NP_, cites_src, cites_dst, EC_,
             W + 0 * (size_t)K * HID_, al + 0 * H_ * D_, ar + 0 * H_ * D_, K, accP);
    // writes: author -> paper (type 1)
    gat_type(haIn, NA_, hpIn, NP_, writes_src, writes_dst, EW_,
             W + 1 * (size_t)K * HID_, al + 1 * H_ * D_, ar + 1 * H_ * D_, K, accP);
    // written-by: paper -> author (type 2)
    gat_type(hpIn, NP_, haIn, NA_, wb_src, wb_dst, EB_,
             W + 2 * (size_t)K * HID_, al + 2 * H_ * D_, ar + 2 * H_ * D_, K, accA);

    if (l < 2) {  // ReLU into next-layer feature buffers
      relu_copy_k<<<cdiv((long long)NP_ * HID_, 256), 256, 0, stream>>>(accP, hpB,
                                                                        (size_t)NP_ * HID_);
      relu_copy_k<<<cdiv((long long)NA_ * HID_, 256), 256, 0, stream>>>(accA, haB,
                                                                        (size_t)NA_ * HID_);
    }
  }

  // classifier: out = accP @ linW + linb   [NP x NCLS], fused bias
  gemm(accP, linW, (float*)d_out, NP_, NCLS_, HID_, linb);
}
